// CrossAttention_8873402433681
// MI455X (gfx1250) — matrix operand order repaired and checked by reading, not compiled, not run on hardware
//
#include <hip/hip_runtime.h>

// MI455X / gfx1250 cross-attention, bf16 WMMA with f32 accumulation.
// Workspace layout (assumes ws_size >= 24 MiB):
//   Qbf [4096][1024] bf16   (8 MiB)
//   Kbf [4096][1024] bf16   (8 MiB)
//   Vt  [1024][4096] bf16   (8 MiB, transposed V for P@V B-fragments)

typedef __bf16 bf16_t;
typedef __attribute__((ext_vector_type(16))) __bf16 v16bf;
typedef __attribute__((ext_vector_type(8)))  __bf16 v8bf;
typedef __attribute__((ext_vector_type(8)))  float  v8f;
typedef __attribute__((ext_vector_type(4)))  float  v4f;

union V16U { v16bf v; v8bf h[2]; };

#define SEQ1 4096
#define SEQ2 4096
#define DMODEL 1024   // D_IN == D_KQ == D_V == 1024

// ---------------------------------------------------------------------------
// Projection GEMM: Y(bf16) = X(f32)[M=4096, K=1024] @ W(f32)[K=1024, N=1024]
// Block tile 64(M) x 128(N), K-step 32, double-buffered LDS staging so the
// next tile's global loads overlap the current tile's 4 WMMAs per wave.
// TRANSPOSE=1: store Y[col][row] (V -> Vt) as packed b128 stores.
// TRANSPOSE=0: stage C tile in LDS, then coalesced b128 stores.
// ---------------------------------------------------------------------------
#define XS_STRIDE 40   // 80B rows: 16B-aligned fragments, conflict-free banks
#define XS_SZ (64 * XS_STRIDE)
#define WT_SZ (128 * XS_STRIDE)

template <int TRANSPOSE>
__global__ __launch_bounds__(256) void proj_gemm_bf16(
    const float* __restrict__ X, const float* __restrict__ W,
    bf16_t* __restrict__ Y)
{
  __shared__ bf16_t Xs[2][XS_SZ];
  __shared__ bf16_t Wt[2][WT_SZ];   // W tile stored transposed: [col][k]
  __shared__ bf16_t Cs[64 * 128];   // epilogue staging (row-major output)

  const int tid  = threadIdx.x;
  const int lane = tid & 31;
  const int wid  = tid >> 5;
  const int rowg = wid >> 1;        // 0..3
  const int colg = wid & 1;         // 0..1
  const int m0   = blockIdx.x * 64;
  const int n0   = blockIdx.y * 128;
  const int ln   = lane & 15;

  // staging coordinates (fixed per thread)
  const int xr = tid >> 2, xc = (tid & 3) * 8;       // X: 64x32, 8 elems/thread
  const int wr = tid >> 3, wc = (tid & 7) * 16;      // W: 32x128, 16 elems/thread

  v8f acc[4] = {};

  // stage k0 = 0 into buffer 0
  {
    const float* sx = X + (size_t)(m0 + xr) * DMODEL + xc;
    #pragma unroll
    for (int i = 0; i < 8; ++i) Xs[0][xr * XS_STRIDE + xc + i] = (bf16_t)sx[i];
    const float* sw = W + (size_t)wr * DMODEL + n0 + wc;
    #pragma unroll
    for (int i = 0; i < 16; ++i) Wt[0][(wc + i) * XS_STRIDE + wr] = (bf16_t)sw[i];
  }
  __syncthreads();

  for (int k0 = 0; k0 < DMODEL; k0 += 32) {
    const int buf = (k0 >> 5) & 1;
    // stage next tile into the other buffer (overlaps with WMMAs below)
    if (k0 + 32 < DMODEL) {
      const float* sx = X + (size_t)(m0 + xr) * DMODEL + (k0 + 32) + xc;
      #pragma unroll
      for (int i = 0; i < 8; ++i)
        Xs[buf ^ 1][xr * XS_STRIDE + xc + i] = (bf16_t)sx[i];
      const float* sw = W + (size_t)(k0 + 32 + wr) * DMODEL + n0 + wc;
      #pragma unroll
      for (int i = 0; i < 16; ++i)
        Wt[buf ^ 1][(wc + i) * XS_STRIDE + wr] = (bf16_t)sw[i];
    }

    // A fragment: 16x32 bf16 (ISA layout: lane<16 -> K{0..7,16..23})
    V16U a;
    {
      int row = rowg * 16 + ln;
      int kb  = (lane < 16) ? 0 : 8;
      a.h[0] = *(const v8bf*)&Xs[buf][row * XS_STRIDE + kb];
      a.h[1] = *(const v8bf*)&Xs[buf][row * XS_STRIDE + 16 + kb];
    }
    #pragma unroll
    for (int j = 0; j < 4; ++j) {
      V16U b;                       // B fragment: 32x16, lane holds column
      int col = colg * 64 + j * 16 + ln;
      int kb  = (lane < 16) ? 0 : 8;
      b.h[0] = *(const v8bf*)&Wt[buf][col * XS_STRIDE + kb];
      b.h[1] = *(const v8bf*)&Wt[buf][col * XS_STRIDE + kb + 16];
      acc[j] = __builtin_amdgcn_wmma_f32_16x16x32_bf16(
          false, a.v, false, b.v, (short)0, acc[j], false, false);
    }
    __syncthreads();   // one barrier per k-step (double-buffered)
  }

  // C layout: VGPR r, lanes 0-15 -> M=r, lanes 16-31 -> M=r+8, N=lane&15
  if (TRANSPOSE) {
    // Rows are contiguous in Vt: pack 8 bf16 -> one global_store_b128.
    #pragma unroll
    for (int j = 0; j < 4; ++j) {
      int col   = n0 + colg * 64 + j * 16 + ln;
      int rbase = m0 + rowg * 16 + ((lane < 16) ? 0 : 8);
      v8bf pk;
      #pragma unroll
      for (int r = 0; r < 8; ++r) pk[r] = (bf16_t)acc[j][r];
      *(v8bf*)&Y[(size_t)col * SEQ2 + rbase] = pk;
    }
  } else {
    // Stage through LDS, then fully coalesced b128 stores.
    #pragma unroll
    for (int j = 0; j < 4; ++j) {
      int colL   = colg * 64 + j * 16 + ln;
      int rbaseL = rowg * 16 + ((lane < 16) ? 0 : 8);
      #pragma unroll
      for (int r = 0; r < 8; ++r)
        Cs[(rbaseL + r) * 128 + colL] = (bf16_t)acc[j][r];
    }
    __syncthreads();
    {
      int row = tid >> 2;            // 0..63
      int c0  = (tid & 3) * 32;      // 0,32,64,96
      const v8bf* src = (const v8bf*)&Cs[row * 128 + c0];
      v8bf* dst = (v8bf*)&Y[(size_t)(m0 + row) * DMODEL + n0 + c0];
      #pragma unroll
      for (int i = 0; i < 4; ++i) dst[i] = src[i];
    }
  }
}

// ---------------------------------------------------------------------------
// Flash attention, single pass, O accumulator in LDS (CDNA5: 320KB/WGP).
// Block: 32 query rows, 8 waves = (rowg 0..1) x (seg 0..3).
// Key tiles of 64. Per tile: S (16x16/wave, 32 WMMAs), online softmax
// via cross-lane reduction + LDS stats, P->LDS bf16, O += P@V (32 WMMAs).
// ---------------------------------------------------------------------------
#define OSTRIDE 1028   // 32x1028 f32 O accumulator (padded vs 64-bank LDS)
#define PSTRIDE 72     // 32x72 bf16 P tile (144B rows: 16B aligned frags)

__global__ __launch_bounds__(256) void flash_attn(
    const bf16_t* __restrict__ Q, const bf16_t* __restrict__ K,
    const bf16_t* __restrict__ Vt, float* __restrict__ Out)
{
  extern __shared__ char smem[];
  float*  O_s    = (float*)smem;                                   // 32*1028*4
  bf16_t* P_s    = (bf16_t*)(smem + 32 * OSTRIDE * 4);             // 32*72*2
  float*  pmax_s = (float*)(smem + 32 * OSTRIDE * 4 + 32 * PSTRIDE * 2); // 8*16
  float*  psum_s = pmax_s + 128;
  float*  m_s    = psum_s + 128;                                   // 32
  float*  l_s    = m_s + 32;                                       // 32

  const int tid   = threadIdx.x;
  const int lane  = tid & 31;
  const int wid   = tid >> 5;
  const int rowg  = wid >> 2;          // 0..1  (16 query rows each)
  const int seg   = wid & 3;           // 0..3  (16 keys / 256 V cols each)
  const int q0    = blockIdx.x * 32;
  const int half8 = (lane >> 4) * 8;   // C-layout row offset per lane half
  const int ln    = lane & 15;
  const int akb   = (lane < 16) ? 0 : 8;
  const int bkb   = (lane < 16) ? 0 : 8;
  const float scale = 0.03125f;        // 1/sqrt(1024)

  for (int i = tid; i < 32 * OSTRIDE; i += 256) O_s[i] = 0.0f;
  if (tid < 32) { m_s[tid] = -1e30f; l_s[tid] = 0.0f; }
  __syncthreads();

  const bf16_t* qp = Q + (size_t)(q0 + rowg * 16 + ln) * DMODEL;

  for (int kt = 0; kt < SEQ2 / 64; ++kt) {
    const int key0 = kt * 64;

    // ---- S tile: 16 rows x 16 keys per wave, reduce over D_KQ ----
    v8f sacc = {};
    const bf16_t* kp = K + (size_t)(key0 + seg * 16 + ln) * DMODEL;
    if (kt + 1 < SEQ2 / 64) {
      // overlap next K tile's L2 fill with this tile's 32-WMMA reduction
      __builtin_prefetch(kp + (size_t)DMODEL * 64, 0, 1);
    }
    for (int ks = 0; ks < DMODEL; ks += 32) {
      V16U a, b;
      a.h[0] = *(const v8bf*)(qp + ks + akb);
      a.h[1] = *(const v8bf*)(qp + ks + 16 + akb);
      b.h[0] = *(const v8bf*)(kp + ks + bkb);
      b.h[1] = *(const v8bf*)(kp + ks + bkb + 16);
      sacc = __builtin_amdgcn_wmma_f32_16x16x32_bf16(
          false, a.v, false, b.v, (short)0, sacc, false, false);
    }

    // ---- partial row max over this wave's 16 keys ----
    float s[8], rmax[8];
    #pragma unroll
    for (int r = 0; r < 8; ++r) {
      s[r] = sacc[r] * scale;
      float v = s[r];
      #pragma unroll
      for (int msk = 1; msk < 16; msk <<= 1) v = fmaxf(v, __shfl_xor(v, msk, 32));
      rmax[r] = v;
    }
    if (ln == 0) {
      #pragma unroll
      for (int r = 0; r < 8; ++r)
        pmax_s[(rowg * 4 + seg) * 16 + half8 + r] = rmax[r];
    }
    __syncthreads();   // A: pmax visible

    // ---- combine maxima, correction factors (per-lane, for its 8 rows) ----
    float mnew[8], corr[8];
    #pragma unroll
    for (int r = 0; r < 8; ++r) {
      int row = rowg * 16 + half8 + r;
      float mo = m_s[row];
      float mn = mo;
      #pragma unroll
      for (int sg = 0; sg < 4; ++sg)
        mn = fmaxf(mn, pmax_s[(rowg * 4 + sg) * 16 + half8 + r]);
      mnew[r] = mn;
      corr[r] = __expf(mo - mn);
    }

    // ---- P = exp(s - mnew); partial row sums; P -> LDS bf16 ----
    float rsum[8];
    #pragma unroll
    for (int r = 0; r < 8; ++r) {
      float p = __expf(s[r] - mnew[r]);
      P_s[(rowg * 16 + half8 + r) * PSTRIDE + seg * 16 + ln] = (bf16_t)p;
      float v = p;
      #pragma unroll
      for (int msk = 1; msk < 16; msk <<= 1) v += __shfl_xor(v, msk, 32);
      rsum[r] = v;
    }
    if (ln == 0) {
      #pragma unroll
      for (int r = 0; r < 8; ++r)
        psum_s[(rowg * 4 + seg) * 16 + half8 + r] = rsum[r];
    }
    __syncthreads();   // B: P + psum visible

    // ---- stats finalize (one wave per row group) ----
    if (seg == 0 && ln == 0) {
      #pragma unroll
      for (int r = 0; r < 8; ++r) {
        int row = rowg * 16 + half8 + r;
        float lacc = l_s[row] * corr[r];
        #pragma unroll
        for (int sg = 0; sg < 4; ++sg)
          lacc += psum_s[(rowg * 4 + sg) * 16 + half8 + r];
        l_s[row] = lacc;
        m_s[row] = mnew[r];
      }
    }

    // ---- O(16 rows x 256 cols per wave) = O*corr + P @ V ----
    V16U pa[2];
    {
      int prow = rowg * 16 + ln;
      #pragma unroll
      for (int kc = 0; kc < 2; ++kc) {
        pa[kc].h[0] = *(const v8bf*)&P_s[prow * PSTRIDE + kc * 32 + akb];
        pa[kc].h[1] = *(const v8bf*)&P_s[prow * PSTRIDE + kc * 32 + 16 + akb];
      }
    }
    #pragma unroll 4
    for (int t = 0; t < 16; ++t) {
      int col = seg * 256 + t * 16 + ln;
      float* obase = O_s + col;
      v8f oacc;
      #pragma unroll
      for (int r = 0; r < 8; ++r)
        oacc[r] = obase[(rowg * 16 + half8 + r) * OSTRIDE] * corr[r];
      #pragma unroll
      for (int kc = 0; kc < 2; ++kc) {
        V16U b;  // B 32x16 chunk: lane holds V column, keys contiguous in Vt
        const bf16_t* vp = Vt + (size_t)col * SEQ2 + key0 + kc * 32 + bkb;
        b.h[0] = *(const v8bf*)(vp);
        b.h[1] = *(const v8bf*)(vp + 16);
        oacc = __builtin_amdgcn_wmma_f32_16x16x32_bf16(
            false, pa[kc].v, false, b.v, (short)0, oacc, false, false);
      }
      #pragma unroll
      for (int r = 0; r < 8; ++r)
        obase[(rowg * 16 + half8 + r) * OSTRIDE] = oacc[r];
    }
    __syncthreads();   // C: O/P/stats stable before next tile
  }

  // ---- normalize by l, write f32 output ----
  {
    int row = tid >> 3;                 // 0..31
    int c0  = (tid & 7) * 128;
    float inv_l = 1.0f / l_s[row];
    const float* src = O_s + row * OSTRIDE + c0;
    float* dst = Out + (size_t)(q0 + row) * DMODEL + c0;
    for (int c = 0; c < 128; c += 4) {
      v4f v = *(const v4f*)(src + c);
      v *= inv_l;
      *(v4f*)(dst + c) = v;
    }
  }
}

// ---------------------------------------------------------------------------
extern "C" void kernel_launch(void* const* d_in, const int* in_sizes, int n_in,
                              void* d_out, int out_size, void* d_ws, size_t ws_size,
                              hipStream_t stream) {
  const float* x1 = (const float*)d_in[0];
  const float* x2 = (const float*)d_in[1];
  const float* Wq = (const float*)d_in[2];
  const float* Wk = (const float*)d_in[3];
  const float* Wv = (const float*)d_in[4];
  float* out = (float*)d_out;

  bf16_t* Qbf = (bf16_t*)d_ws;
  bf16_t* Kbf = Qbf + (size_t)SEQ1 * DMODEL;
  bf16_t* Vt  = Kbf + (size_t)SEQ2 * DMODEL;

  dim3 pgrid(SEQ1 / 64, DMODEL / 128);
  proj_gemm_bf16<0><<<pgrid, 256, 0, stream>>>(x1, Wq, Qbf);
  proj_gemm_bf16<0><<<pgrid, 256, 0, stream>>>(x2, Wk, Kbf);
  proj_gemm_bf16<1><<<pgrid, 256, 0, stream>>>(x2, Wv, Vt);   // transposed V

  size_t smem = (size_t)32 * OSTRIDE * 4 + (size_t)32 * PSTRIDE * 2
              + 128 * 4 + 128 * 4 + 32 * 4 + 32 * 4;          // 137472 B
  flash_attn<<<SEQ1 / 32, 256, smem, stream>>>(Qbf, Kbf, Vt, out);
}
